// MultiHeadAttention_new_87814901334217
// MI455X (gfx1250) — matrix operand order repaired and checked by reading, not compiled, not run on hardware
//
#include <hip/hip_runtime.h>
#include <math.h>

typedef __attribute__((ext_vector_type(16))) _Float16 v16h;
typedef __attribute__((ext_vector_type(8)))  _Float16 v8h;
typedef __attribute__((ext_vector_type(4)))  _Float16 v4h;
typedef __attribute__((ext_vector_type(8)))  float    v8f;
typedef __attribute__((ext_vector_type(4)))  float    v4f;
typedef __attribute__((ext_vector_type(4)))  unsigned int v4u;
typedef __attribute__((ext_vector_type(8)))  int      v8i;
typedef __attribute__((ext_vector_type(4)))  int      v4i;

#define BSZ    2048
#define DIN    1024
#define DMODEL 1024
#define NH     16
#define DK     64
#define S_STRIDE 2052   // 2048 + 4 float pad: bank-conflict-free, keeps 16B align

// LDS map for attn_kernel (bytes):
//   S    : 16*S_STRIDE f32                      [0, 131328)
//   part : 8*16*16 f32 (also reused as Q tile)  [131328, 139520)
//   red  : 16*16 f32
//   rowstat : 16 f32
#define LDS_PART_OFF (16 * S_STRIDE * 4)

#define HAS_TDM (__has_builtin(__builtin_amdgcn_tensor_load_to_lds) && \
                 __has_builtin(__builtin_amdgcn_s_wait_tensorcnt))

__device__ __forceinline__ v8f wmma16(v16h a, v16h b, v8f c) {
    // D = A(16x32 f16) x B(32x16 f16) + C(16x16 f32)
    return __builtin_amdgcn_wmma_f32_16x16x32_f16(false, a, false, b, (short)0, c,
                                                  false, false);
}

// A fragment (16x32 f16) from contiguous f32 row; element e <-> k = e + 16*hi
__device__ __forceinline__ v16h packA_f32(const float* row, int kb, int hi) {
    const v4f* p0 = (const v4f*)(row + kb + 16 * hi);
    const v4f* p1 = (const v4f*)(row + kb + 16 * hi + 8);
    v4f r0 = p0[0], r1 = p0[1], r2 = p1[0], r3 = p1[1];
    v16h a;
#pragma unroll
    for (int i = 0; i < 4; ++i) {
        a[i]      = (_Float16)r0[i];
        a[4 + i]  = (_Float16)r1[i];
        a[8 + i]  = (_Float16)r2[i];
        a[12 + i] = (_Float16)r3[i];
    }
    return a;
}

template <typename P>
__device__ __forceinline__ v16h packA_f16(const P* row, int kb, int hi) {
    v8h r0 = *(const v8h*)(row + kb + 16 * hi);
    v8h r1 = *(const v8h*)(row + kb + 16 * hi + 8);
    v16h a;
#pragma unroll
    for (int i = 0; i < 8; ++i) { a[i] = r0[i]; a[8 + i] = r1[i]; }
    return a;
}

// B fragment (32x16 f16): lane = column, element e <-> k = e + 16*hi (contiguous run)
__device__ __forceinline__ v16h packB_f32(const float* colrun) {
    const v4f* q = (const v4f*)colrun;
    v4f w0 = q[0], w1 = q[1], w2 = q[2], w3 = q[3];
    v16h b;
#pragma unroll
    for (int i = 0; i < 4; ++i) {
        b[i]      = (_Float16)w0[i];
        b[4 + i]  = (_Float16)w1[i];
        b[8 + i]  = (_Float16)w2[i];
        b[12 + i] = (_Float16)w3[i];
    }
    return b;
}

__device__ __forceinline__ v16h packB_f16(const _Float16* colrun) {
    v8h r0 = ((const v8h*)colrun)[0];
    v8h r1 = ((const v8h*)colrun)[1];
    v16h b;
#pragma unroll
    for (int i = 0; i < 8; ++i) { b[i] = r0[i]; b[8 + i] = r1[i]; }
    return b;
}

// ---------------------------------------------------------------------------
// Kernel 1: projection GEMM  C[m,n] = X[m,:].W[n,:] + b[n] -> f16
// MODE 0: out[h][m][dk] (value*scale);  MODE 1: out[h][dk][m] (V transposed)
// ---------------------------------------------------------------------------
template <int MODE>
__global__ __launch_bounds__(256) void proj_kernel(const float* __restrict__ X,
                                                   const float* __restrict__ W,
                                                   const float* __restrict__ bias,
                                                   _Float16* __restrict__ out,
                                                   float scale) {
    const int lane = threadIdx.x & 31;
    const int wave = threadIdx.x >> 5;
    const int hi = lane >> 4;
    const int lr = lane & 15;
    const int m0 = blockIdx.y * 128 + wave * 16;
    const int n0 = blockIdx.x * 64;

    const float* Arow = X + (size_t)(m0 + lr) * DIN;

    v8f acc[4] = {};
    for (int kb = 0; kb < DIN; kb += 32) {
        v16h a = packA_f32(Arow, kb, hi);
#pragma unroll
        for (int nt = 0; nt < 4; ++nt) {
            const int n = n0 + nt * 16 + lr;
            const float* wrun = W + (size_t)n * DIN + kb + 16 * hi;
            if (kb + 32 < DIN) __builtin_prefetch(wrun + 32, 0, 0);
            acc[nt] = wmma16(a, packB_f32(wrun), acc[nt]);
        }
    }

#pragma unroll
    for (int nt = 0; nt < 4; ++nt) {
        const int n = n0 + nt * 16 + lr;
        const float bn = bias[n];
        const int h = n >> 6, dk = n & 63;
#pragma unroll
        for (int r = 0; r < 8; ++r) {
            const int m = m0 + r + 8 * hi;
            const float v = (acc[nt][r] + bn) * scale;
            if (MODE == 0)
                out[((size_t)h * BSZ + m) * DK + dk] = (_Float16)v;
            else
                out[((size_t)h * DK + dk) * BSZ + m] = (_Float16)v;
        }
    }
}

// ---------------------------------------------------------------------------
// Kernel 2: fused attention for one (head, 16-query block).
//   Q tile -> LDS via TDM (tensor_load_to_lds, s_wait_tensorcnt)
//   scores (WMMA) -> LDS slab [16][2048] f32
//   softmax in LDS (v_exp_f32, float4 passes), attn written once (b128 stores)
//   P.V: wave (kh,dkt) owns dk tile dkt*16.. and keys kh*1024.., LDS reduce
// LDS = 140.6 KB -> 2 workgroups per 320KB WGP (4 waves/SIMD).
// ---------------------------------------------------------------------------
__global__ __launch_bounds__(256) void attn_kernel(const _Float16* __restrict__ Qh,
                                                   const _Float16* __restrict__ Kh,
                                                   const _Float16* __restrict__ Vt,
                                                   float* __restrict__ attn,
                                                   _Float16* __restrict__ concat) {
    extern __shared__ float smem[];
    float* S       = smem;                    // [16][S_STRIDE]
    float* part    = S + 16 * S_STRIDE;       // [8][16][16] (first 2KB doubles as Q tile)
    float* red     = part + 8 * 16 * 16;      // [16][16]
    float* rowstat = red + 16 * 16;           // [16]

    const int lane = threadIdx.x & 31;
    const int wave = threadIdx.x >> 5;
    const int hi = lane >> 4;
    const int lr = lane & 15;
    const int h = blockIdx.y;
    const int m0 = blockIdx.x * 16;

    const _Float16* Qbase = Qh + (size_t)h * BSZ * DK;
    const _Float16* Kbase = Kh + (size_t)h * BSZ * DK;
    const _Float16* Vbase = Vt + (size_t)h * DK * BSZ;

    // ---- Q tile (16x64 f16, contiguous rows): TDM -> LDS, else direct ----
    v16h aq[2];
#if HAS_TDM
    if (wave == 0) {
        const unsigned long long ga =
            (unsigned long long)(size_t)(Qbase + (size_t)m0 * DK);
        v4u g0 = {};
        g0[0] = 1u;                                    // count=1 valid descriptor
        g0[1] = (unsigned)LDS_PART_OFF;                // lds_addr (bytes)
        g0[2] = (unsigned)(ga & 0xffffffffull);        // global_addr[31:0]
        g0[3] = (unsigned)((ga >> 32) & 0x1ffffffull)  // global_addr[56:32]
                | (2u << 30);                          // type = 2 ("image")
        v8i g1 = {};
        g1[0] = (int)(1u << 16);                       // data_size = 1 -> 2 bytes
        g1[1] = (int)(64u << 16);                      // tensor_dim0 = 64
        g1[2] = (int)(2048u << 16);                    // tensor_dim1 = 2048
        g1[3] = (int)(64u << 16);                      // tile_dim0 = 64
        g1[4] = 16;                                    // tile_dim1 = 16
        g1[5] = 64;                                    // tensor_dim0_stride = 64
        v4i gz = {};
#if defined(__clang_major__) && __clang_major__ >= 23
        v8i gz8 = {};
        __builtin_amdgcn_tensor_load_to_lds(g0, g1, gz, gz, gz8, 0);
#else
        __builtin_amdgcn_tensor_load_to_lds(g0, g1, gz, gz, 0);
#endif
        __builtin_amdgcn_s_wait_tensorcnt(0);
    }
    __syncthreads();
    {
        const _Float16* Qlds = (const _Float16*)part + lr * DK;
#pragma unroll
        for (int kb = 0; kb < 2; ++kb) aq[kb] = packA_f16(Qlds, kb * 32, hi);
    }
    __syncthreads();   // Q consumed into registers before part is reused
#else
#pragma unroll
    for (int kb = 0; kb < 2; ++kb)
        aq[kb] = packA_f16(Qbase + (size_t)(m0 + lr) * DK, kb * 32, hi);
#endif

    // ---- scores: wave owns keys [wave*256, +256) ----
    for (int nt = 0; nt < 16; ++nt) {
        const int nbase = wave * 256 + nt * 16;
        v8f acc = {};
#pragma unroll
        for (int kb = 0; kb < 2; ++kb) {
            const _Float16* krun =
                Kbase + (size_t)(nbase + lr) * DK + kb * 32 + 16 * hi;
            acc = wmma16(aq[kb], packB_f16(krun), acc);
        }
#pragma unroll
        for (int r = 0; r < 8; ++r)
            S[(r + 8 * hi) * S_STRIDE + nbase + lr] = acc[r];
    }

    // ---- warm up this wave's P.V working set of V while softmax runs ----
    {
        const int dkt = wave & 3;          // dk tile this wave will compute
        const int kh = wave >> 2;          // key half this wave will contract
#pragma unroll
        for (int it = 0; it < 8; ++it) {
            const int unit = it * 32 + lane;           // 256 cachelines of 128B
            const int vrow = dkt * 16 + (unit >> 4);
            const int seg = unit & 15;
            __builtin_prefetch(Vbase + (size_t)vrow * BSZ + kh * 1024 + seg * 64,
                               0, 0);
        }
    }
    __syncthreads();

    // ---- softmax: 16 threads per row, float4 passes ----
    const int row = threadIdx.x >> 4;
    const int tr = threadIdx.x & 15;
    float* Srow = S + row * S_STRIDE;

    float mx = -3.4e38f;
    for (int j = tr * 4; j < BSZ; j += 64) {
        v4f x = *(const v4f*)(Srow + j);
        mx = fmaxf(mx, fmaxf(fmaxf(x[0], x[1]), fmaxf(x[2], x[3])));
    }
    red[row * 16 + tr] = mx;
    __syncthreads();
    if (tr == 0) {
        float m = red[row * 16];
#pragma unroll
        for (int i = 1; i < 16; ++i) m = fmaxf(m, red[row * 16 + i]);
        rowstat[row] = m;
    }
    __syncthreads();
    mx = rowstat[row];

    float sum = 0.f;
    for (int j = tr * 4; j < BSZ; j += 64) {
        v4f x = *(const v4f*)(Srow + j);
        v4f e;
#pragma unroll
        for (int i = 0; i < 4; ++i) e[i] = __expf(x[i] - mx);
        *(v4f*)(Srow + j) = e;
        sum += (e[0] + e[1]) + (e[2] + e[3]);
    }
    __syncthreads();
    red[row * 16 + tr] = sum;
    __syncthreads();
    if (tr == 0) {
        float s = 0.f;
#pragma unroll
        for (int i = 0; i < 16; ++i) s += red[row * 16 + i];
        rowstat[row] = 1.0f / s;
    }
    __syncthreads();
    const float inv = rowstat[row];

    float* attn_row = attn + ((size_t)h * BSZ + (m0 + row)) * BSZ;
    for (int j = tr * 4; j < BSZ; j += 64) {
        v4f x = *(const v4f*)(Srow + j);
        v4f p = {x[0] * inv, x[1] * inv, x[2] * inv, x[3] * inv};
        *(v4f*)(Srow + j) = p;              // normalized P stays in LDS for P.V
        *(v4f*)(attn_row + j) = p;          // 268 MB: the bandwidth floor
    }
    __syncthreads();

    // ---- P.V: wave = kh*4 + dkt; dk tile [dkt*16,+16), keys [kh*1024,+1024) ----
    const int dkt = wave & 3;
    const int kh = wave >> 2;
    v8f oacc = {};
#pragma unroll 4
    for (int ks = 0; ks < 32; ++ks) {
        const int kbase = kh * 1024 + ks * 32;
        // A from LDS P (f32 -> f16); A row = lr
        const float* prow = S + lr * S_STRIDE + kbase;
        const v4f* p0 = (const v4f*)(prow + 16 * hi);
        const v4f* p1 = (const v4f*)(prow + 16 * hi + 8);
        v4f r0 = p0[0], r1 = p0[1], r2 = p1[0], r3 = p1[1];
        v16h af;
#pragma unroll
        for (int i = 0; i < 4; ++i) {
            af[i]      = (_Float16)r0[i];
            af[4 + i]  = (_Float16)r1[i];
            af[8 + i]  = (_Float16)r2[i];
            af[12 + i] = (_Float16)r3[i];
        }
        const _Float16* vrun =
            Vbase + (size_t)(dkt * 16 + lr) * BSZ + kbase + 16 * hi;
        oacc = wmma16(af, packB_f16(vrun), oacc);
    }

#pragma unroll
    for (int r = 0; r < 8; ++r)
        part[(wave * 16 + r + 8 * hi) * 16 + lr] = oacc[r];
    __syncthreads();

    // ---- reduce the two key-halves, write concat as f16x4 ----
    {
        const int t4 = threadIdx.x * 4;          // 1024 outputs, 4 per thread
        const int rr = t4 >> 6;
        const int dd0 = t4 & 63;
        const int tl = dd0 >> 4;
        const int c0 = dd0 & 15;
        v4h o;
#pragma unroll
        for (int i = 0; i < 4; ++i) {
            const float s = part[(tl * 16 + rr) * 16 + c0 + i] +
                            part[((tl + 4) * 16 + rr) * 16 + c0 + i];
            o[i] = (_Float16)s;
        }
        *(v4h*)(concat + (size_t)(m0 + rr) * DMODEL + h * 64 + dd0) = o;
    }
}

// ---------------------------------------------------------------------------
// Kernel 3: output GEMM  out[m,n] = concat_f16[m,:].Wo[n,:] + bo[n]  (f32 out)
// ---------------------------------------------------------------------------
__global__ __launch_bounds__(256) void outproj_kernel(const _Float16* __restrict__ A,
                                                      const float* __restrict__ W,
                                                      const float* __restrict__ bias,
                                                      float* __restrict__ out) {
    const int lane = threadIdx.x & 31;
    const int wave = threadIdx.x >> 5;
    const int hi = lane >> 4;
    const int lr = lane & 15;
    const int m0 = blockIdx.y * 128 + wave * 16;
    const int n0 = blockIdx.x * 64;

    const _Float16* Arow = A + (size_t)(m0 + lr) * DMODEL;

    v8f acc[4] = {};
    for (int kb = 0; kb < DMODEL; kb += 32) {
        v16h a = packA_f16(Arow, kb, hi);
#pragma unroll
        for (int nt = 0; nt < 4; ++nt) {
            const int n = n0 + nt * 16 + lr;
            const float* wrun = W + (size_t)n * DMODEL + kb + 16 * hi;
            if (kb + 32 < DMODEL) __builtin_prefetch(wrun + 32, 0, 0);
            acc[nt] = wmma16(a, packB_f32(wrun), acc[nt]);
        }
    }

#pragma unroll
    for (int nt = 0; nt < 4; ++nt) {
        const int n = n0 + nt * 16 + lr;
        const float bn = bias[n];
#pragma unroll
        for (int r = 0; r < 8; ++r)
            out[(size_t)(m0 + r + 8 * hi) * DMODEL + n] = acc[nt][r] + bn;
    }
}

extern "C" void kernel_launch(void* const* d_in, const int* in_sizes, int n_in,
                              void* d_out, int out_size, void* d_ws, size_t ws_size,
                              hipStream_t stream) {
    const float* q  = (const float*)d_in[0];
    const float* k  = (const float*)d_in[1];
    const float* v  = (const float*)d_in[2];
    const float* Wq = (const float*)d_in[3];
    const float* bq = (const float*)d_in[4];
    const float* Wk = (const float*)d_in[5];
    const float* bk = (const float*)d_in[6];
    const float* Wv = (const float*)d_in[7];
    const float* bv = (const float*)d_in[8];
    const float* Wo = (const float*)d_in[9];
    const float* bo = (const float*)d_in[10];

    float* out  = (float*)d_out;                     // [BSZ][DMODEL]
    float* attn = out + (size_t)BSZ * DMODEL;        // [NH][BSZ][BSZ]

    _Float16* Qh     = (_Float16*)d_ws;              // [NH][BSZ][DK] (pre-scaled)
    _Float16* Kh     = Qh + (size_t)NH * BSZ * DK;   // [NH][BSZ][DK]
    _Float16* Vt     = Kh + (size_t)NH * BSZ * DK;   // [NH][DK][BSZ] (transposed)
    _Float16* concat = Vt + (size_t)NH * BSZ * DK;   // [BSZ][DMODEL]

    const dim3 gproj(DMODEL / 64, BSZ / 128);
    proj_kernel<0><<<gproj, 256, 0, stream>>>(q, Wq, bq, Qh, 0.125f); // 1/sqrt(64)
    proj_kernel<0><<<gproj, 256, 0, stream>>>(k, Wk, bk, Kh, 1.0f);
    proj_kernel<1><<<gproj, 256, 0, stream>>>(v, Wv, bv, Vt, 1.0f);

    const size_t smem =
        (size_t)(16 * S_STRIDE + 8 * 16 * 16 + 16 * 16 + 16) * sizeof(float);
    attn_kernel<<<dim3(BSZ / 16, NH), 256, smem, stream>>>(Qh, Kh, Vt, attn, concat);

    outproj_kernel<<<gproj, 256, 0, stream>>>(concat, Wo, bo, out);
}
